// AttentionHead_24275155157643
// MI455X (gfx1250) — matrix operand order repaired and checked by reading, not compiled, not run on hardware
//
#include <hip/hip_runtime.h>

typedef __attribute__((ext_vector_type(16))) _Float16     v16h;
typedef __attribute__((ext_vector_type(8)))  float        v8f;
typedef __attribute__((ext_vector_type(4)))  unsigned int v4u;
typedef __attribute__((ext_vector_type(8)))  int          v8i;
typedef __attribute__((ext_vector_type(4)))  int          v4i;

#define D_MODEL 1024
#define D_HEAD  64
#define T_SEQ   4096
#define NBATCH  4

union FragU { v16h h; unsigned int u[8]; };

__device__ __forceinline__ unsigned int pack2h(float f0, float f1) {
    _Float16 h0 = (_Float16)f0, h1 = (_Float16)f1;
    return ((unsigned int)__builtin_bit_cast(unsigned short, h1) << 16) |
            (unsigned int)__builtin_bit_cast(unsigned short, h0);
}

// ---------------------------------------------------------------------------
// TDM: async 2D tile load (global -> LDS), f16 elements.
// Descriptor layout per CDNA5 ISA.
// ---------------------------------------------------------------------------
#if __has_include(<hip/amd_detail/amd_gfx1250_TDM.h>)
#define TDM_CALL(g0, g1, z0, z1) \
    __builtin_amdgcn_tensor_load_to_lds(g0, g1, z0, z1, (v8i){}, 0)
#else
#define TDM_CALL(g0, g1, z0, z1) \
    __builtin_amdgcn_tensor_load_to_lds(g0, g1, z0, z1, 0)
#endif

__device__ __forceinline__ void tdm_load_tile_2d(
        unsigned int lds_addr,            // LDS byte offset
        const _Float16* gptr,             // tile start in global memory
        unsigned int tensor_d0, unsigned int tensor_d1,   // elements
        unsigned int tile_d0,   unsigned int tile_d1,     // elements
        unsigned long long stride0)       // elements between rows
{
    unsigned long long ga = (unsigned long long)(size_t)gptr;
    v4u g0 = {};
    g0[0] = 1u;                                          // count=1, user desc
    g0[1] = lds_addr;                                    // lds_addr[31:0]
    g0[2] = (unsigned int)(ga & 0xFFFFFFFFu);            // global_addr[31:0]
    g0[3] = (unsigned int)((ga >> 32) & 0x01FFFFFFu)     // global_addr[56:32]
          | (2u << 30);                                  // type = 2 ("image")
    v8i g1 = {};
    g1[0] = (int)(1u << 16);                             // data_size=1 (2B)
    g1[1] = (int)((tensor_d0 & 0xFFFFu) << 16);          // tensor_dim0[15:0]
    g1[2] = (int)(((tensor_d0 >> 16) & 0xFFFFu) |
                  ((tensor_d1 & 0xFFFFu) << 16));        // dim0[31:16]|dim1[15:0]
    g1[3] = (int)(((tensor_d1 >> 16) & 0xFFFFu) |
                  ((tile_d0 & 0xFFFFu) << 16));          // dim1[31:16]|tile_dim0
    g1[4] = (int)(tile_d1 & 0xFFFFu);                    // tile_dim1 (tile_dim2=0)
    g1[5] = (int)(stride0 & 0xFFFFFFFFull);              // dim0_stride[31:0]
    g1[6] = (int)((stride0 >> 32) & 0xFFFFull);          // dim0_stride[47:32]
    g1[7] = 0;
    v4i z = {};
    TDM_CALL(g0, g1, z, z);
}

// ---------------------------------------------------------------------------
// Kernel 1: convert + transpose weights: Wt[mat][d][k] = W_mat[k][d] (f16)
// mat order: 0=Q, 1=K, 2=V
// ---------------------------------------------------------------------------
__global__ void wtrans_kernel(const float* __restrict__ Wq,
                              const float* __restrict__ Wk,
                              const float* __restrict__ Wv,
                              _Float16* __restrict__ Wt) {
    int i = blockIdx.x * 256 + threadIdx.x;      // 3*64*1024 total
    int mat = i / (D_HEAD * D_MODEL);
    int rem = i % (D_HEAD * D_MODEL);
    int d = rem / D_MODEL;
    int k = rem % D_MODEL;
    const float* W = (mat == 0) ? Wq : ((mat == 1) ? Wk : Wv);
    Wt[i] = (_Float16)W[k * D_HEAD + d];
}

// ---------------------------------------------------------------------------
// Kernel 2: projections Q/K/V = x @ W via WMMA f16 (f32 accumulate).
// Block = 96 threads (3 waves); wave w computes matrix w for a 16-row tile.
// ---------------------------------------------------------------------------
__global__ __launch_bounds__(96)
void proj_kernel(const float* __restrict__ x, const _Float16* __restrict__ Wt,
                 _Float16* __restrict__ Qh, _Float16* __restrict__ Kh,
                 _Float16* __restrict__ Vt) {
    const int wave = threadIdx.x >> 5;     // 0:Q 1:K 2:V
    const int lane = threadIdx.x & 31;
    const int h    = lane >> 4;
    const int ln   = lane & 15;
    const int row0 = blockIdx.x * 16;      // flattened row in [0, B*T)
    const _Float16* W = Wt + (size_t)wave * D_HEAD * D_MODEL;
    const float* xrow = x + (size_t)(row0 + ln) * D_MODEL;

    v8f acc[4] = {};

    for (int k0 = 0; k0 < D_MODEL; k0 += 32) {
        if (k0 + 32 < D_MODEL)
            __builtin_prefetch(xrow + k0 + 32 + 8 * h, 0, 0);  // next A strip
        // A fragment: x rows (convert f32 -> f16 on the fly)
        FragU a;
#pragma unroll
        for (int r = 0; r < 8; ++r) {
            int kl = 2 * r + 16 * h;
            const float* px = xrow + k0 + kl;
            a.u[r] = pack2h(px[0], px[1]);
        }
        // gather ALL B fragments first, then issue the 4 WMMAs back-to-back
        FragU bf[4];
#pragma unroll
        for (int nt = 0; nt < 4; ++nt)
#pragma unroll
            for (int r = 0; r < 8; ++r) {
                int kl = 2 * r + 16 * h;
                bf[nt].u[r] = *(const unsigned int*)(
                    W + (size_t)(nt * 16 + ln) * D_MODEL + k0 + kl);
            }
#pragma unroll
        for (int nt = 0; nt < 4; ++nt)
            acc[nt] = __builtin_amdgcn_wmma_f32_16x16x32_f16(
                false, a.h, false, bf[nt].h, (short)0, acc[nt], false, false);
    }

    if (wave < 2) {
        _Float16* O = (wave == 0) ? Qh : Kh;
#pragma unroll
        for (int nt = 0; nt < 4; ++nt)
#pragma unroll
            for (int v = 0; v < 8; ++v) {
                int m = v + 8 * h;  // D layout: M = vgpr + 8*half
                O[(size_t)(row0 + m) * D_HEAD + nt * 16 + ln] = (_Float16)acc[nt][v];
            }
    } else {
        // V transposed: Vt[b][d][t]; lane holds fixed d, 8 consecutive t
        int b  = row0 >> 12;
        int t0 = (row0 & (T_SEQ - 1)) + 8 * h;
#pragma unroll
        for (int nt = 0; nt < 4; ++nt) {
            int d = nt * 16 + ln;
            uint4 pk;
            pk.x = pack2h(acc[nt][0], acc[nt][1]);
            pk.y = pack2h(acc[nt][2], acc[nt][3]);
            pk.z = pack2h(acc[nt][4], acc[nt][5]);
            pk.w = pack2h(acc[nt][6], acc[nt][7]);
            *(uint4*)(Vt + ((size_t)b * D_HEAD + d) * T_SEQ + t0) = pk;
        }
    }
}

// ---------------------------------------------------------------------------
// Kernel 3: flash attention, causal. Block = 128 thr (4 waves), 64 q rows.
// K/V tiles staged by the Tensor Data Mover, double-buffered in LDS.
// ---------------------------------------------------------------------------
__global__ __launch_bounds__(128)
void attn_kernel(const _Float16* __restrict__ Qh, const _Float16* __restrict__ Kh,
                 const _Float16* __restrict__ Vt, float* __restrict__ out) {
    __shared__ _Float16 Ks[2][64 * 64];   // [buf][s][d]
    __shared__ _Float16 Vs[2][64 * 64];   // [buf][d][s]

    const int b    = blockIdx.x >> 6;
    const int qt0  = (blockIdx.x & 63) * 64;
    const int wave = threadIdx.x >> 5;
    const int lane = threadIdx.x & 31;
    const int h    = lane >> 4;
    const int ln   = lane & 15;
    const int qw0  = qt0 + 16 * wave;
    const int q_lane = qw0 + ln;

    const _Float16* Kbase = Kh + (size_t)(b * T_SEQ) * D_HEAD;
    const _Float16* Vbase = Vt + (size_t)(b * D_HEAD) * T_SEQ;
    unsigned int ldsK[2] = { (unsigned int)(size_t)&Ks[0][0],
                             (unsigned int)(size_t)&Ks[1][0] };
    unsigned int ldsV[2] = { (unsigned int)(size_t)&Vs[0][0],
                             (unsigned int)(size_t)&Vs[1][0] };

    // Q^T B-fragments (persist across the s loop): B[k=d][n=q]
    FragU qb[2];
#pragma unroll
    for (int df = 0; df < 2; ++df)
#pragma unroll
        for (int r = 0; r < 8; ++r) {
            int d = df * 32 + 16 * h + 2 * r;
            qb[df].u[r] = *(const unsigned int*)(
                Qh + ((size_t)(b * T_SEQ) + qw0 + ln) * D_HEAD + d);
        }

    const int niter = (qt0 >> 6) + 1;

    // preload tile 0 into buffer 0 (wave 0 drives the TDM)
    if (wave == 0) {
        tdm_load_tile_2d(ldsK[0], Kbase, 64, T_SEQ, 64, 64, 64);
        tdm_load_tile_2d(ldsV[0], Vbase, T_SEQ, 64, 64, 64, T_SEQ);
    }

    float mstate = -INFINITY, lstate = 0.0f;
    v8f acc[4] = {};

    for (int i = 0; i < niter; ++i) {
        const int s0  = i << 6;
        const int cur = i & 1;
        __syncthreads();   // everyone done reading buf[cur^1] from iter i-1
        if (wave == 0) {
            if (i + 1 < niter) {
                int s1 = (i + 1) << 6;
                tdm_load_tile_2d(ldsK[cur ^ 1], Kbase + (size_t)s1 * 64,
                                 64, T_SEQ, 64, 64, 64);
                tdm_load_tile_2d(ldsV[cur ^ 1], Vbase + s1,
                                 T_SEQ, 64, 64, 64, T_SEQ);
                __builtin_amdgcn_s_wait_tensorcnt(2);  // pair i done; pair i+1 in flight
            } else {
                __builtin_amdgcn_s_wait_tensorcnt(0);
            }
        }
        __syncthreads();   // buf[cur] ready for all waves
        const _Float16* Kc = &Ks[cur][0];
        const _Float16* Vc = &Vs[cur][0];

        // --- St = K @ Q^T : 4 subtiles of 16 s-rows ---
        v8f st[4];
#pragma unroll
        for (int t = 0; t < 4; ++t) {
            v8f c = {};
#pragma unroll
            for (int df = 0; df < 2; ++df) {
                FragU ka;  // A[m=s][k=d] from LDS (d contiguous)
#pragma unroll
                for (int r = 0; r < 8; ++r) {
                    int d = df * 32 + 16 * h + 2 * r;
                    ka.u[r] = *(const unsigned int*)(Kc + (t * 16 + ln) * 64 + d);
                }
                c = __builtin_amdgcn_wmma_f32_16x16x32_f16(
                    false, ka.h, false, qb[df].h, (short)0, c, false, false);
            }
            st[t] = c;
        }

        // --- scale, causal mask, online softmax (rows are per-lane) ---
        float tmax = -INFINITY;
#pragma unroll
        for (int t = 0; t < 4; ++t)
#pragma unroll
            for (int v = 0; v < 8; ++v) {
                int sg = s0 + t * 16 + v + 8 * h;        // St row = s
                float val = st[t][v] * 0.125f;           // d_head^-0.5
                val = (sg > q_lane) ? -INFINITY : val;
                st[t][v] = val;
                tmax = fmaxf(tmax, val);
            }
        tmax = fmaxf(tmax, __shfl_xor(tmax, 16, 32));
        float mnew  = fmaxf(mstate, tmax);
        float alpha = __expf(mstate - mnew);
        float rsum = 0.0f;
#pragma unroll
        for (int t = 0; t < 4; ++t)
#pragma unroll
            for (int v = 0; v < 8; ++v) {
                float p = __expf(st[t][v] - mnew);
                st[t][v] = p;
                rsum += p;
            }
        rsum += __shfl_xor(rsum, 16, 32);
        lstate = lstate * alpha + rsum;
        mstate = mnew;

        // --- rescale accumulator (scale lives in lane q; acc rows in vgprs) ---
#pragma unroll
        for (int v = 0; v < 8; ++v) {
            float av = __shfl(alpha, v + 8 * h, 32);
#pragma unroll
            for (int nt = 0; nt < 4; ++nt) acc[nt][v] *= av;
        }

        // --- build P A-fragments directly from St layout ---
        FragU pa[2];
#pragma unroll
        for (int sf = 0; sf < 2; ++sf)
#pragma unroll
            for (int r = 0; r < 8; ++r) {
                int t = 2 * sf + (r >= 4);
                int v = (r < 4) ? 2 * r : 2 * (r - 4);
                pa[sf].u[r] = pack2h(st[t][v], st[t][v + 1]);
            }

        // --- acc += P @ V ---
#pragma unroll
        for (int nt = 0; nt < 4; ++nt)
#pragma unroll
            for (int sf = 0; sf < 2; ++sf) {
                FragU vb;  // B[k=s][n=d] from Vs[d][s] (s contiguous)
#pragma unroll
                for (int r = 0; r < 8; ++r) {
                    int sl = sf * 32 + ((r < 4) ? (2 * r + 8 * h) : (16 + 2 * (r - 4) + 8 * h));
                    vb.u[r] = *(const unsigned int*)(Vc + (nt * 16 + ln) * 64 + sl);
                }
                acc[nt] = __builtin_amdgcn_wmma_f32_16x16x32_f16(
                    false, pa[sf].h, false, vb.h, (short)0, acc[nt], false, false);
            }
    }

    // --- epilogue: divide by row sum, store f32 ---
#pragma unroll
    for (int v = 0; v < 8; ++v) {
        float lv  = __shfl(lstate, v + 8 * h, 32);
        float inv = 1.0f / lv;
        int q = qw0 + v + 8 * h;
#pragma unroll
        for (int nt = 0; nt < 4; ++nt)
            out[((size_t)(b * T_SEQ) + q) * D_HEAD + nt * 16 + ln] = acc[nt][v] * inv;
    }
}

// ---------------------------------------------------------------------------
extern "C" void kernel_launch(void* const* d_in, const int* in_sizes, int n_in,
                              void* d_out, int out_size, void* d_ws, size_t ws_size,
                              hipStream_t stream) {
    const float* x  = (const float*)d_in[0];
    const float* Wk = (const float*)d_in[1];
    const float* Wq = (const float*)d_in[2];
    const float* Wv = (const float*)d_in[3];
    float* out = (float*)d_out;

    _Float16* ws = (_Float16*)d_ws;
    _Float16* Wt = ws;                                             // 3*64*1024
    _Float16* Qh = ws + (size_t)3 * D_HEAD * D_MODEL;              // 4*4096*64
    _Float16* Kh = Qh + (size_t)NBATCH * T_SEQ * D_HEAD;
    _Float16* Vt = Kh + (size_t)NBATCH * T_SEQ * D_HEAD;

    wtrans_kernel<<<(3 * D_HEAD * D_MODEL) / 256, 256, 0, stream>>>(Wq, Wk, Wv, Wt);
    proj_kernel<<<(NBATCH * T_SEQ) / 16, 96, 0, stream>>>(x, Wt, Qh, Kh, Vt);
    attn_kernel<<<NBATCH * (T_SEQ / 64), 128, 0, stream>>>(Qh, Kh, Vt, out);
}
